// VectorAttention_21741124452508
// MI455X (gfx1250) — matrix operand order repaired and checked by reading, not compiled, not run on hardware
//
#include <hip/hip_runtime.h>
#include <math.h>

typedef _Float16 v8h  __attribute__((ext_vector_type(8)));
typedef _Float16 v16h __attribute__((ext_vector_type(16)));
typedef float    v8f  __attribute__((ext_vector_type(8)));

#define BATCH   8
#define NPT     256
#define DIM     64
#define ITILE   8     // i rows per block (one per wave)
#define JTILE   16    // j cols per block (rows of the WMMA M-tile)
#define WSTRIDE 72    // padded row stride in halves (16B pad -> conflict-free b128)
#define VSTRIDE 68    // padded row stride in floats

union HFrag { v16h v; v8h h[2]; };

// Fused pair pipeline. ACCUM=false: emit raw scores. ACCUM=true: recompute
// joined, apply softmax weights from stats, accumulate output.
template<bool ACCUM>
__global__ __launch_bounds__(256)
void va_pair_kernel(const float* __restrict__ positions,
                    const float* __restrict__ values,
                    const float* __restrict__ Wv1, const float* __restrict__ bv1,
                    const float* __restrict__ Wv2, const float* __restrict__ bv2,
                    const float* __restrict__ Ws1, const float* __restrict__ bs1,
                    const float* __restrict__ Ws2, const float* __restrict__ bs2,
                    float* __restrict__ scores,
                    const float* __restrict__ stats,
                    float* __restrict__ out)
{
    __shared__ _Float16 ldsWv2[DIM * WSTRIDE];   // f16 B-operand, [N][K] row-major padded
    __shared__ _Float16 ldsWs1[DIM * WSTRIDE];
    __shared__ float    ldsVi[ITILE * VSTRIDE];  // values rows for this block's i-tile
    __shared__ float    ldsVj[JTILE * VSTRIDE];  // values rows for this block's j-tile
    __shared__ float    ldsWv1[2 * DIM];
    __shared__ float    ldsBv1[DIM], ldsBv2[DIM], ldsBs1[DIM], ldsWs2[DIM];
    __shared__ _Float16 ldsH[8 * 16 * WSTRIDE];  // per-wave 16x64 f16 A-operand staging
    __shared__ float    ldsOut[DIM];

    const int tid = threadIdx.x;
    const int jt  = blockIdx.x & 15;        // 16 j-tiles
    const int it  = (blockIdx.x >> 4) & 31; // 32 i-tiles
    const int b   = blockIdx.x >> 9;        // 8 batches
    const int i0  = it * ITILE;
    const int j0  = jt * JTILE;

    // ---- cooperative staging (f32 -> f16 for WMMA weights) ----
    for (int e = tid; e < DIM * DIM; e += 256) {
        const int r = e >> 6, c = e & 63;
        ldsWv2[c * WSTRIDE + r] = (_Float16)Wv2[e];
        ldsWs1[c * WSTRIDE + r] = (_Float16)Ws1[e];
    }
    for (int e = tid; e < ITILE * DIM; e += 256) {
        const int r = e >> 6, c = e & 63;
        ldsVi[r * VSTRIDE + c] = values[((size_t)(b * NPT) + i0 + r) * DIM + c];
    }
    for (int e = tid; e < JTILE * DIM; e += 256) {
        const int r = e >> 6, c = e & 63;
        ldsVj[r * VSTRIDE + c] = values[((size_t)(b * NPT) + j0 + r) * DIM + c];
    }
    if (tid < 2 * DIM) ldsWv1[tid] = Wv1[tid];
    if (tid < DIM) {
        ldsBv1[tid] = bv1[tid]; ldsBv2[tid] = bv2[tid];
        ldsBs1[tid] = bs1[tid]; ldsWs2[tid] = Ws2[tid];
        if (ACCUM) ldsOut[tid] = 0.0f;
    }
    __syncthreads();

    const int wv   = tid >> 5;   // wave id 0..7 -> i = i0 + wv
    const int lane = tid & 31;
    const int hi   = lane >> 4;  // half-wave select
    const int m    = lane & 15;  // N-lane / row index within half
    const int i    = i0 + wv;

    // ---- geometric invariants: pair (i, j0+m), rows of the M-tile = j offset ----
    const float* pi = positions + ((size_t)(b * NPT) + i) * 3;
    const float* pj = positions + ((size_t)(b * NPT) + j0 + m) * 3;
    const float pix = pi[0], piy = pi[1], piz = pi[2];
    const float pjx = pj[0], pjy = pj[1], pjz = pj[2];
    const float dotv = pix * pjx + piy * pjy + piz * pjz;
    const float wx = piy * pjz - piz * pjy;
    const float wy = piz * pjx - pix * pjz;
    const float wz = pix * pjy - piy * pjx;
    const float bn = sqrtf(wx * wx + wy * wy + wz * wz + 1e-20f);

    _Float16* hb = &ldsH[wv * 16 * WSTRIDE];

    // ---- value-MLP layer 1 (K=2: VALU, not worth WMMA), row r = j offset ----
    #pragma unroll 8
    for (int k = 0; k < 32; ++k) {
        const int e = lane + 32 * k;
        const int r = e >> 6, c = e & 63;
        const float d0 = __shfl(dotv, r, 32);
        const float d1 = __shfl(bn,   r, 32);
        const float h  = fmaxf(d0 * ldsWv1[c] + d1 * ldsWv1[DIM + c] + ldsBv1[c], 0.0f);
        hb[r * WSTRIDE + c] = (_Float16)h;
    }

    // ---- value-MLP layer 2: (16x64)x(64x64) via 8 WMMAs ----
    v8f acc[4];
    #pragma unroll
    for (int nt = 0; nt < 4; ++nt) acc[nt] = (v8f){0, 0, 0, 0, 0, 0, 0, 0};

    #pragma unroll
    for (int kc = 0; kc < 2; ++kc) {
        HFrag a;  // A 16x32 f16 layout: lane=M, halves = K {0-7,16-23} | {8-15,24-31}
        a.h[0] = *(const v8h*)&hb[m * WSTRIDE + kc * 32 + hi * 8];
        a.h[1] = *(const v8h*)&hb[m * WSTRIDE + kc * 32 + 16 + hi * 8];
        #pragma unroll
        for (int nt = 0; nt < 4; ++nt) {
            HFrag bf; // B 32x16 f16 layout: lane=N, halves = K {0-7,16-23} | {8-15,24-31}
            const _Float16* wp = &ldsWv2[(nt * 16 + m) * WSTRIDE + kc * 32 + hi * 8];
            bf.h[0] = *(const v8h*)wp;
            bf.h[1] = *(const v8h*)(wp + 16);
            acc[nt] = __builtin_amdgcn_wmma_f32_16x16x32_f16(
                false, a.v, false, bf.v, (short)0, acc[nt], false, false);
        }
    }

    // ---- joined = 0.5*(invar_values + 0.5*(v_i + v_j)); restage as f16 A ----
    #pragma unroll
    for (int nt = 0; nt < 4; ++nt) {
        const int n = nt * 16 + m;
        const float vin = ldsVi[wv * VSTRIDE + n];
        #pragma unroll
        for (int r = 0; r < 8; ++r) {
            const int M = r + 8 * hi;  // D-layout: VGPR r, half hi -> row M
            const float vjn = ldsVj[M * VSTRIDE + n];
            const float jv  = 0.5f * (acc[nt][r] + ldsBv2[n] + 0.5f * (vin + vjn));
            acc[nt][r] = jv;
            hb[M * WSTRIDE + n] = (_Float16)jv;
        }
    }

    // ---- score-MLP layer 1: (16x64)x(64x64) via 8 WMMAs ----
    v8f sacc[4];
    #pragma unroll
    for (int nt = 0; nt < 4; ++nt) sacc[nt] = (v8f){0, 0, 0, 0, 0, 0, 0, 0};

    #pragma unroll
    for (int kc = 0; kc < 2; ++kc) {
        HFrag a;
        a.h[0] = *(const v8h*)&hb[m * WSTRIDE + kc * 32 + hi * 8];
        a.h[1] = *(const v8h*)&hb[m * WSTRIDE + kc * 32 + 16 + hi * 8];
        #pragma unroll
        for (int nt = 0; nt < 4; ++nt) {
            HFrag bf;
            const _Float16* wp = &ldsWs1[(nt * 16 + m) * WSTRIDE + kc * 32 + hi * 8];
            bf.h[0] = *(const v8h*)wp;
            bf.h[1] = *(const v8h*)(wp + 16);
            sacc[nt] = __builtin_amdgcn_wmma_f32_16x16x32_f16(
                false, a.v, false, bf.v, (short)0, sacc[nt], false, false);
        }
    }

    // ---- score = relu(h2) . Ws2 + bs2 : per-lane partials + 16-lane xor reduce ----
    float part[8];
    #pragma unroll
    for (int r = 0; r < 8; ++r) part[r] = 0.0f;
    #pragma unroll
    for (int nt = 0; nt < 4; ++nt) {
        const int n = nt * 16 + m;
        const float w2  = ldsWs2[n];
        const float bsn = ldsBs1[n];
        #pragma unroll
        for (int r = 0; r < 8; ++r)
            part[r] += fmaxf(sacc[nt][r] + bsn, 0.0f) * w2;
    }
    #pragma unroll
    for (int r = 0; r < 8; ++r) {
        part[r] += __shfl_xor(part[r], 1, 32);
        part[r] += __shfl_xor(part[r], 2, 32);
        part[r] += __shfl_xor(part[r], 4, 32);
        part[r] += __shfl_xor(part[r], 8, 32);
    }
    const float bs2v = bs2[0];

    if (!ACCUM) {
        if (m == 0) {  // lanes 0 and 16 each own 8 rows
            #pragma unroll
            for (int r = 0; r < 8; ++r) {
                const int M = r + 8 * hi;
                scores[((size_t)(b * NPT) + i) * NPT + j0 + M] = part[r] + bs2v;
            }
        }
    } else {
        const float mx   = stats[b * 2 + 0];
        const float invs = stats[b * 2 + 1];
        float aw[8];
        #pragma unroll
        for (int r = 0; r < 8; ++r)
            aw[r] = expf(part[r] + bs2v - mx) * invs;
        #pragma unroll
        for (int nt = 0; nt < 4; ++nt) {
            float v = 0.0f;
            #pragma unroll
            for (int r = 0; r < 8; ++r) v += aw[r] * acc[nt][r];
            atomicAdd(&ldsOut[nt * 16 + m], v);
        }
        __syncthreads();
        if (tid < DIM) atomicAdd(&out[b * DIM + tid], ldsOut[tid]);
    }
}

// Per-batch softmax statistics (max, 1/sum-exp) + zero the output accumulator.
__global__ __launch_bounds__(256)
void va_stats_kernel(const float* __restrict__ scores,
                     float* __restrict__ stats,
                     float* __restrict__ out)
{
    __shared__ float red[256];
    const int b   = blockIdx.x;
    const int tid = threadIdx.x;
    const float* s = scores + (size_t)b * NPT * NPT;

    float mv = -INFINITY;
    for (int idx = tid; idx < NPT * NPT; idx += 256) mv = fmaxf(mv, s[idx]);
    red[tid] = mv;
    __syncthreads();
    for (int o = 128; o > 0; o >>= 1) {
        if (tid < o) red[tid] = fmaxf(red[tid], red[tid + o]);
        __syncthreads();
    }
    const float mx = red[0];
    __syncthreads();

    float sum = 0.0f;
    for (int idx = tid; idx < NPT * NPT; idx += 256) sum += expf(s[idx] - mx);
    red[tid] = sum;
    __syncthreads();
    for (int o = 128; o > 0; o >>= 1) {
        if (tid < o) red[tid] += red[tid + o];
        __syncthreads();
    }
    if (tid == 0) {
        stats[b * 2 + 0] = mx;
        stats[b * 2 + 1] = 1.0f / red[0];
    }
    if (tid < DIM) out[b * DIM + tid] = 0.0f;
}

extern "C" void kernel_launch(void* const* d_in, const int* in_sizes, int n_in,
                              void* d_out, int out_size, void* d_ws, size_t ws_size,
                              hipStream_t stream) {
    const float* positions = (const float*)d_in[0];
    const float* values    = (const float*)d_in[1];
    const float* Wv1 = (const float*)d_in[2];
    const float* bv1 = (const float*)d_in[3];
    const float* Wv2 = (const float*)d_in[4];
    const float* bv2 = (const float*)d_in[5];
    const float* Ws1 = (const float*)d_in[6];
    const float* bs1 = (const float*)d_in[7];
    const float* Ws2 = (const float*)d_in[8];
    const float* bs2 = (const float*)d_in[9];
    float* out = (float*)d_out;

    float* scores = (float*)d_ws;                       // 8*65536 floats = 2 MB
    float* stats  = scores + (size_t)BATCH * NPT * NPT; // 16 floats

    const int nblocks = BATCH * (NPT / ITILE) * (NPT / JTILE); // 4096

    va_pair_kernel<false><<<nblocks, 256, 0, stream>>>(
        positions, values, Wv1, bv1, Wv2, bv2, Ws1, bs1, Ws2, bs2,
        scores, nullptr, nullptr);

    va_stats_kernel<<<BATCH, 256, 0, stream>>>(scores, stats, out);

    va_pair_kernel<true><<<nblocks, 256, 0, stream>>>(
        positions, values, Wv1, bv1, Wv2, bv2, Ws1, bs1, Ws2, bs2,
        scores, stats, out);
}
